// SimpleGCN_14199161880828
// MI455X (gfx1250) — matrix operand order repaired and checked by reading, not compiled, not run on hardware
//
#include <hip/hip_runtime.h>
#include <hip/hip_bf16.h>

typedef __attribute__((ext_vector_type(16))) _Float16 v16h;
typedef __attribute__((ext_vector_type(8)))  float    v8f;

#define N_NODES  50000
#define NPAD     50048   // 64 * 782, exact tiling for gin_mlp
#define N_EDGES  800000
#define IN_DIM   128
#define HID      128
#define OUT_DIM  64
#define N_GRAPHS 512

// LDS row stride in f16 elements: 136 = 68 words -> row r starts at bank
// (r*68)%64 = 4r, spreading the 16 A-rows across all 64 banks (conflict-free
// ds_load_b128 fragment loads), while keeping 16B per-lane contiguity.
#define LSTR 136

// ---------------------------------------------------------------- weights
// Pack row-major weight W[rows x 128] (h = x @ W^T) into WMMA B fragments.
// Fragment layout (16-bit B, 32x16 per chunk): lane L holds column N = L%16,
// element e holds K = 32*chunk + (e&7) + 16*(e>>3) + 8*(L/16).
// P[((t*4 + c)*32 + L)*16 + e] = (f16) W[16t + L%16][32c + (e&7) + 16*(e>>3) + 8*(L>>4)]
__global__ void pack_w(const float* __restrict__ W, _Float16* __restrict__ P,
                       int n_tiles) {
  int i = blockIdx.x * blockDim.x + threadIdx.x;
  int total = n_tiles * 4 * 32 * 16;
  if (i >= total) return;
  int e    = i & 15;
  int lane = (i >> 4) & 31;
  int c    = (i >> 9) & 3;
  int t    = i >> 11;
  int row  = 16 * t + (lane & 15);
  int col  = 32 * c + (e & 7) + 16 * (e >> 3) + 8 * (lane >> 4);
  P[i] = (_Float16)W[row * 128 + col];
}

// ---------------------------------------------- embedding (dual write)
// Writes x into BOTH the gather-source buffer and the aggregation buffer,
// so edge atomics accumulate directly on top of x (GIN self term, eps=0).
// Pads rows [N_NODES, NPAD) with zeros so downstream tiling is exact.
__global__ void embed2(const float* __restrict__ emb, const int* __restrict__ idx,
                       float* __restrict__ x, float* __restrict__ xa) {
  int i = blockIdx.x * blockDim.x + threadIdx.x;
  if (i >= NPAD * 128) return;
  int n = i >> 7, k = i & 127;
  float v = (n < N_NODES) ? emb[idx[n] * IN_DIM + k] : 0.0f;
  x[i]  = v;
  xa[i] = v;
}

// --------------------------------------------------- edge scatter-add
// 32 lanes per edge; wave-uniform edge index (scalarized via readfirstlane);
// each lane moves 4 contiguous floats. agg buffer is pre-seeded with x.
__global__ void edge_agg(const float* __restrict__ x, const int* __restrict__ src,
                         const int* __restrict__ dst, float* __restrict__ agg) {
  int t = blockIdx.x * blockDim.x + threadIdx.x;
  int e = t >> 5;
  if (e >= N_EDGES) return;
  int lane = t & 31;
  int s = __builtin_amdgcn_readfirstlane(src[e]);
  int d = __builtin_amdgcn_readfirstlane(dst[e]);
  float4 v = ((const float4*)(x + (size_t)s * 128))[lane];
  float* ad = agg + (size_t)d * 128 + lane * 4;
  atomicAdd(ad + 0, v.x);
  atomicAdd(ad + 1, v.y);
  atomicAdd(ad + 2, v.z);
  atomicAdd(ad + 3, v.w);
}

// --------------------------------------------------------- fused GIN MLP
// One workgroup (8 waves) per 64-node tile. Wave w owns output features
// [16w, 16w+16); B fragments loaded once and reused across 4 m-tiles.
// Two chained 128x128 matmuls via v_wmma_f32_16x16x32_f16 (f32 accumulate),
// ReLU after both. Result written to two buffers: the next layer's gather
// source and the next layer's aggregation base (GIN self term).
__global__ __launch_bounds__(256)
void gin_mlp(const float* __restrict__ xin,
             const _Float16* __restrict__ pA, const float* __restrict__ bA,
             const _Float16* __restrict__ pB, const float* __restrict__ bB,
             float* __restrict__ out1, float* __restrict__ out2) {
  __shared__ _Float16 ldsX[64 * LSTR];   // ~17 KB, bank-conflict-free stride
  __shared__ _Float16 ldsH[64 * LSTR];   // ~17 KB
  const int tid = threadIdx.x;
  const size_t base = (size_t)blockIdx.x * 64 * 128;

  // stage input tile -> LDS f16 (padded row stride)
  for (int i = tid; i < 64 * 128; i += 256) {
    int m = i >> 7, k = i & 127;
    ldsX[m * LSTR + k] = (_Float16)xin[base + i];
  }
  __syncthreads();

  const int w    = tid >> 5;
  const int lane = tid & 31;
  const int hl   = lane >> 4;               // wave half
  const int mloc = lane & 15;               // A-row within m-tile
  const int j    = (w << 4) + (lane & 15);  // C/D column = output feature
  const float biasA = bA[j];
  const float biasB = bB[j];

  v16h bf[4];

  // ---- matmul 1: H = relu(X @ A^T + bA)
#pragma unroll
  for (int c = 0; c < 4; ++c)
    bf[c] = *(const v16h*)(pA + (((w * 4 + c) * 32 + lane) << 4));
#pragma unroll
  for (int mt = 0; mt < 4; ++mt) {
    v8f acc = {};
#pragma unroll
    for (int c = 0; c < 4; ++c) {
      v16h a;
      const _Float16* xr = ldsX + (mt * 16 + mloc) * LSTR + c * 32 + 8 * hl;
#pragma unroll
      for (int e = 0; e < 16; ++e)          // ISA 16-bit A layout
        a[e] = xr[(e & 7) + ((e >> 3) << 4)];
      acc = __builtin_amdgcn_wmma_f32_16x16x32_f16(false, a, false, bf[c],
                                                   (short)0, acc, false, false);
    }
#pragma unroll
    for (int r = 0; r < 8; ++r) {           // C/D: M = r + 8*hl, N = j
      float v = acc[r] + biasA;
      v = v > 0.0f ? v : 0.0f;
      ldsH[(mt * 16 + r + 8 * hl) * LSTR + j] = (_Float16)v;
    }
  }
  __syncthreads();

  // ---- matmul 2: out = relu(H @ B^T + bB)
#pragma unroll
  for (int c = 0; c < 4; ++c)
    bf[c] = *(const v16h*)(pB + (((w * 4 + c) * 32 + lane) << 4));
#pragma unroll
  for (int mt = 0; mt < 4; ++mt) {
    v8f acc = {};
#pragma unroll
    for (int c = 0; c < 4; ++c) {
      v16h a;
      const _Float16* hr = ldsH + (mt * 16 + mloc) * LSTR + c * 32 + 8 * hl;
#pragma unroll
      for (int e = 0; e < 16; ++e)
        a[e] = hr[(e & 7) + ((e >> 3) << 4)];
      acc = __builtin_amdgcn_wmma_f32_16x16x32_f16(false, a, false, bf[c],
                                                   (short)0, acc, false, false);
    }
#pragma unroll
    for (int r = 0; r < 8; ++r) {
      float v = acc[r] + biasB;
      v = v > 0.0f ? v : 0.0f;
      size_t o = base + (size_t)(mt * 16 + r + 8 * hl) * 128 + j;
      out1[o] = v;
      out2[o] = v;
    }
  }
}

// ------------------------------------------------- segment mean pool
// batch is sorted: one block per graph, binary-search the node range,
// coalesced column-sum. No atomics, divide fused.
__device__ __forceinline__ int lbound(const int* __restrict__ a, int n, int key) {
  int lo = 0, hi = n;
  while (lo < hi) {
    int mid = (lo + hi) >> 1;
    if (a[mid] < key) lo = mid + 1; else hi = mid;
  }
  return lo;
}

__global__ __launch_bounds__(128)
void pool_seg(const float* __restrict__ x, const int* __restrict__ batch,
              float* __restrict__ pooled) {
  const int g = blockIdx.x;
  const int tid = threadIdx.x;     // one feature column per thread
  const int lo = lbound(batch, N_NODES, g);
  const int hi = lbound(batch, N_NODES, g + 1);
  float s = 0.0f;
  for (int n = lo; n < hi; ++n)
    s += x[(size_t)n * 128 + tid];
  float c = (hi > lo) ? (float)(hi - lo) : 1.0f;
  pooled[(size_t)g * 128 + tid] = s / c;
}

// -------------------------------------------------------------- final FC
// One workgroup (4 waves) per 16-graph tile: [16 x 128] @ [128 x 64] WMMA.
__global__ __launch_bounds__(128)
void fc_k(const float* __restrict__ pooled, const _Float16* __restrict__ pW,
          const float* __restrict__ bias, float* __restrict__ out) {
  __shared__ _Float16 ldsX[16 * LSTR];
  const int tid = threadIdx.x;
  const size_t g0 = (size_t)blockIdx.x * 16;
  for (int i = tid; i < 16 * 128; i += 128) {
    int m = i >> 7, k = i & 127;
    ldsX[m * LSTR + k] = (_Float16)pooled[g0 * 128 + i];
  }
  __syncthreads();

  const int w    = tid >> 5;
  const int lane = tid & 31;
  const int hl   = lane >> 4;
  const int mloc = lane & 15;
  const int j    = (w << 4) + (lane & 15);

  v8f acc = {};
#pragma unroll
  for (int c = 0; c < 4; ++c) {
    v16h a, b;
    const _Float16* xr = ldsX + mloc * LSTR + c * 32 + 8 * hl;
#pragma unroll
    for (int e = 0; e < 16; ++e)
      a[e] = xr[(e & 7) + ((e >> 3) << 4)];
    b = *(const v16h*)(pW + (((w * 4 + c) * 32 + lane) << 4));
    acc = __builtin_amdgcn_wmma_f32_16x16x32_f16(false, a, false, b,
                                                 (short)0, acc, false, false);
  }
  float bb = bias[j];
#pragma unroll
  for (int r = 0; r < 8; ++r)
    out[(g0 + r + 8 * hl) * OUT_DIM + j] = acc[r] + bb;
}

// ---------------------------------------------------------------- launch
extern "C" void kernel_launch(void* const* d_in, const int* in_sizes, int n_in,
                              void* d_out, int out_size, void* d_ws, size_t ws_size,
                              hipStream_t stream) {
  (void)in_sizes; (void)n_in; (void)out_size; (void)ws_size;
  const float* emb = (const float*)d_in[0];
  const float* w1a = (const float*)d_in[1];
  const float* b1a = (const float*)d_in[2];
  const float* w1b = (const float*)d_in[3];
  const float* b1b = (const float*)d_in[4];
  const float* w2a = (const float*)d_in[5];
  const float* b2a = (const float*)d_in[6];
  const float* w2b = (const float*)d_in[7];
  const float* b2b = (const float*)d_in[8];
  const float* fcw = (const float*)d_in[9];
  const float* fcb = (const float*)d_in[10];
  const int* xidx  = (const int*)d_in[11];
  const int* eidx  = (const int*)d_in[12];
  const int* batch = (const int*)d_in[13];
  const int* src = eidx;
  const int* dst = eidx + N_EDGES;

  // workspace carve-out (256B aligned); ~77 MB total
  char* ws = (char*)d_ws;
  size_t off = 0;
  auto carve = [&](size_t bytes) -> char* {
    char* p = ws + off;
    off += (bytes + 255) & ~(size_t)255;
    return p;
  };
  float*    bufX   = (float*)carve((size_t)NPAD * 128 * sizeof(float));
  float*    bufA   = (float*)carve((size_t)NPAD * 128 * sizeof(float));
  float*    bufY   = (float*)carve((size_t)NPAD * 128 * sizeof(float));
  float*    pooled = (float*)carve((size_t)N_GRAPHS * 128 * sizeof(float));
  _Float16* pk1a = (_Float16*)carve(8 * 4 * 32 * 16 * sizeof(_Float16));
  _Float16* pk1b = (_Float16*)carve(8 * 4 * 32 * 16 * sizeof(_Float16));
  _Float16* pk2a = (_Float16*)carve(8 * 4 * 32 * 16 * sizeof(_Float16));
  _Float16* pk2b = (_Float16*)carve(8 * 4 * 32 * 16 * sizeof(_Float16));
  _Float16* pkfc = (_Float16*)carve(4 * 4 * 32 * 16 * sizeof(_Float16));

  const int NF = NPAD * 128;     // 6,406,144
  const int ET = N_EDGES * 32;   // 25,600,000 lane-tasks

  // weight packing (tiny)
  pack_w<<<(8 * 2048 + 255) / 256, 256, 0, stream>>>(w1a, pk1a, 8);
  pack_w<<<(8 * 2048 + 255) / 256, 256, 0, stream>>>(w1b, pk1b, 8);
  pack_w<<<(8 * 2048 + 255) / 256, 256, 0, stream>>>(w2a, pk2a, 8);
  pack_w<<<(8 * 2048 + 255) / 256, 256, 0, stream>>>(w2b, pk2b, 8);
  pack_w<<<(4 * 2048 + 255) / 256, 256, 0, stream>>>(fcw, pkfc, 4);

  // embedding: seeds gather source (bufX) AND aggregation base (bufA)
  embed2<<<(NF + 255) / 256, 256, 0, stream>>>(emb, xidx, bufX, bufA);

  // conv 1: bufA += sum of neighbor rows of bufX; MLP -> bufY (+ agg base bufA)
  edge_agg<<<(ET + 255) / 256, 256, 0, stream>>>(bufX, src, dst, bufA);
  gin_mlp<<<NPAD / 64, 256, 0, stream>>>(bufA, pk1a, b1a, pk1b, b1b, bufY, bufA);

  // conv 2: bufA (= conv1 out) += neighbors of bufY; MLP -> bufX
  edge_agg<<<(ET + 255) / 256, 256, 0, stream>>>(bufY, src, dst, bufA);
  gin_mlp<<<NPAD / 64, 256, 0, stream>>>(bufA, pk2a, b2a, pk2b, b2b, bufX, bufY);

  // sorted-segment mean pool + FC
  pool_seg<<<N_GRAPHS, 128, 0, stream>>>(bufX, batch, pooled);
  fc_k<<<N_GRAPHS / 16, 128, 0, stream>>>(pooled, pkfc, fcb, (float*)d_out);
}
